// ModalityProtoGenerator_23819888623655
// MI455X (gfx1250) — matrix operand order repaired and checked by reading, not compiled, not run on hardware
//
#include <hip/hip_runtime.h>

// MI455X / gfx1250, wave32.
//  - bf16 WMMA (v_wmma_f32_16x16x32_bf16) for both projection GEMMs,
//    fp32 accumulation (bf16 input rounding ~0.2% RMS on unit-variance data).
//  - Async global->LDS DMA (global_load_async_to_lds_b128 / s_wait_asynccnt)
//    with double-buffered tiles so the DMA runs one k-step ahead of the WMMAs.
//  - Attention aggregation collapsed algebraically: output is a head+node mean,
//    so only softmax COLUMN SUMS g[b,h,j] are needed, and
//    exp(leaky_relu(d_i+s_j)) is branch-separable -> 4*512 exps per (b,h).

typedef __bf16 bf16_t;
typedef __attribute__((ext_vector_type(16))) __bf16 v16bf;
typedef __attribute__((ext_vector_type(8)))  float  v8f;

// ---------------------------------------------------------------------------
// CDNA5 async global->LDS copy (ASYNCcnt path). 16B per lane per instruction.
// lds_off = wave-relative LDS byte address (low 32 bits of generic pointer).
// ---------------------------------------------------------------------------
__device__ __forceinline__ void async_b128(unsigned lds_off, const bf16_t* g) {
  asm volatile("global_load_async_to_lds_b128 %0, %1, off"
               :: "v"(lds_off), "v"(g) : "memory");
}
template <int N>
__device__ __forceinline__ void wait_async() {
  asm volatile("s_wait_asynccnt %0" :: "n"(N) : "memory");
}

// ---------------------------------------------------------------------------
// Pre-pass: f32 -> bf16 elementwise (for x).
// ---------------------------------------------------------------------------
__global__ __launch_bounds__(256) void convert_bf16(
    const float* __restrict__ in, bf16_t* __restrict__ out, long long n) {
  long long i = ((long long)blockIdx.x * 256 + threadIdx.x) * 8;
  if (i + 8 <= n) {
#pragma unroll
    for (int e = 0; e < 8; ++e) out[i + e] = (bf16_t)in[i + e];
  }
}

// ---------------------------------------------------------------------------
// Pre-pass: f32 [K,N] -> bf16 [N,K] tiled transpose (weights).
// ---------------------------------------------------------------------------
__global__ __launch_bounds__(256) void transpose_convert(
    const float* __restrict__ in, bf16_t* __restrict__ out, int K, int N) {
  __shared__ bf16_t tile[64][65];
  const int k0 = blockIdx.y * 64, n0 = blockIdx.x * 64;
  const int ty = threadIdx.x >> 6;  // 0..3
  const int tx = threadIdx.x & 63;  // 0..63
#pragma unroll
  for (int rr = 0; rr < 16; ++rr) {
    const int row = ty + rr * 4;  // k within tile
    tile[row][tx] = (bf16_t)in[(size_t)(k0 + row) * N + n0 + tx];
  }
  __syncthreads();
#pragma unroll
  for (int rr = 0; rr < 16; ++rr) {
    const int row = ty + rr * 4;  // n within tile
    out[(size_t)(n0 + row) * K + k0 + tx] = tile[tx][row];
  }
}

// ---------------------------------------------------------------------------
// C(bf16)[M,N] = A(bf16)[M,K] * Bt(bf16)[N,K]^T (+bias).
// Block tile 128x64, BK=32, 8 waves, wave tile 32x32 (2x2 WMMA).
// Double-buffered LDS filled by async DMA one k-step ahead.
// ---------------------------------------------------------------------------
template <bool HAS_BIAS>
__global__ __launch_bounds__(256) void wmma_gemm(
    const bf16_t* __restrict__ A, const bf16_t* __restrict__ Bt,
    const float* __restrict__ bias, bf16_t* __restrict__ C, int N, int K) {
  constexpr int BM = 128, BN = 64, BK = 32;
  constexpr int LDS_ROW = 40;  // bf16/row (80B): conflict-free 16B fragment reads
  constexpr int A_TILE = BM * LDS_ROW;  // elems
  constexpr int B_TILE = BN * LDS_ROW;
  constexpr int BUF = A_TILE + B_TILE;
  __shared__ bf16_t smem[2 * BUF];

  const int tid  = threadIdx.x;
  const int wid  = tid >> 5;
  const int lane = tid & 31;
  const int wr   = wid & 3;
  const int wc   = wid >> 2;
  const int brow = blockIdx.y * BM;
  const int bcol = blockIdx.x * BN;
  const int half = lane >> 4;
  const int l16  = lane & 15;

  v8f acc[2][2] = {};

  const bf16_t* Ag = A + (size_t)brow * K;   // tile origin, advances by BK
  const bf16_t* Bg = Bt + (size_t)bcol * K;
  const unsigned sbase = (unsigned)(size_t)&smem[0];  // LDS byte address

  // per-thread DMA assignments (3 x b128 per thread per tile)
  const int a_r  = tid >> 1;            // 0..127
  const int a_c  = (tid & 1) * 16;      // 0 / 16 (elems)
  const int b_n  = tid >> 2;            // 0..63
  const int b_c  = (tid & 3) * 8;       // 0/8/16/24 (elems)

  auto stage = [&](int buf, int kstep) {
    const unsigned as = sbase + (unsigned)(buf * BUF) * 2;
    const unsigned bs = as + (unsigned)A_TILE * 2;
    const bf16_t* ag = Ag + (size_t)kstep * BK;
    const bf16_t* bg = Bg + (size_t)kstep * BK;
    async_b128(as + (unsigned)(a_r * LDS_ROW + a_c) * 2,       ag + (size_t)a_r * K + a_c);
    async_b128(as + (unsigned)(a_r * LDS_ROW + a_c + 8) * 2,   ag + (size_t)a_r * K + a_c + 8);
    async_b128(bs + (unsigned)(b_n * LDS_ROW + b_c) * 2,       bg + (size_t)b_n * K + b_c);
  };

  const int nk = K / BK;
  stage(0, 0);  // prologue

  for (int s = 0; s < nk; ++s) {
    const int cur = s & 1;
    const bool more = (s + 1) < nk;
    if (more) stage(cur ^ 1, s + 1);       // DMA next tile (buffer safe: barrier below)
    if (more) wait_async<3>(); else wait_async<0>();  // in-order => tile s landed
    __syncthreads();                       // collective: whole tile s in LDS

    const bf16_t* As = smem + cur * BUF;
    const bf16_t* Bs = As + A_TILE;

    v16bf afrag[2], bfrag[2];
#pragma unroll
    for (int ti = 0; ti < 2; ++ti) {
      // ISA 16-bit A 16x32 layout: two contiguous 8-elem runs per lane.
      const bf16_t* p = &As[(wr * 32 + ti * 16 + l16) * LDS_ROW];
      *((uint4*)&afrag[ti])     = *(const uint4*)(p + 8 * half);
      *((uint4*)&afrag[ti] + 1) = *(const uint4*)(p + 16 + 8 * half);
    }
#pragma unroll
    for (int tj = 0; tj < 2; ++tj) {
      // B 32x16: lane reads K = 8*half..+7 and 16+8*half..+7 (matches A layout).
      const bf16_t* p = &Bs[(wc * 32 + tj * 16 + l16) * LDS_ROW + 8 * half];
      *((uint4*)&bfrag[tj])     = *(const uint4*)(p);
      *((uint4*)&bfrag[tj] + 1) = *(const uint4*)(p + 16);
    }
#pragma unroll
    for (int ti = 0; ti < 2; ++ti)
#pragma unroll
      for (int tj = 0; tj < 2; ++tj)
        acc[ti][tj] = __builtin_amdgcn_wmma_f32_16x16x32_bf16(
            false, afrag[ti], false, bfrag[tj], (short)0, acc[ti][tj], false, false);
    __syncthreads();  // protect buf[cur] before it is re-staged at step s+2
  }

  // epilogue: +bias, convert, store
#pragma unroll
  for (int ti = 0; ti < 2; ++ti) {
#pragma unroll
    for (int tj = 0; tj < 2; ++tj) {
      const int col = bcol + wc * 32 + tj * 16 + l16;
      float bv = 0.0f;
      if constexpr (HAS_BIAS) bv = bias[col];
#pragma unroll
      for (int r = 0; r < 8; ++r) {
        const int row = brow + wr * 32 + ti * 16 + half * 8 + r;  // C/D layout
        C[(size_t)row * N + col] = (bf16_t)(acc[ti][tj][r] + bv);
      }
    }
  }
}

// ---------------------------------------------------------------------------
// a_src[bn,h] = <h[bn,h,:], att_src[h,:]>, same for a_dst. One wave per head.
// ---------------------------------------------------------------------------
__global__ __launch_bounds__(256) void attn_logits(
    const bf16_t* __restrict__ h, const float* __restrict__ att_src,
    const float* __restrict__ att_dst, float* __restrict__ a_src,
    float* __restrict__ a_dst) {
  const int bn = blockIdx.x;
  const int w = threadIdx.x >> 5;
  const int lane = threadIdx.x & 31;
  const bf16_t* hp = h + (size_t)bn * 2048 + w * 256;
  const float* as = att_src + w * 256;
  const float* ad = att_dst + w * 256;
  float s = 0.f, d = 0.f;
#pragma unroll
  for (int c = lane; c < 256; c += 32) {
    float hv = (float)hp[c];
    s += hv * as[c];
    d += hv * ad[c];
  }
#pragma unroll
  for (int m = 16; m >= 1; m >>= 1) {
    s += __shfl_xor(s, m, 32);
    d += __shfl_xor(d, m, 32);
  }
  if (lane == 0) {
    a_src[bn * 8 + w] = s;
    a_dst[bn * 8 + w] = d;
  }
}

// ---------------------------------------------------------------------------
// g[b,h,j] = sum_i alpha[b,h,i,j] via branch-separable exp(leaky_relu()).
// ---------------------------------------------------------------------------
__global__ __launch_bounds__(256) void softmax_colsum(
    const float* __restrict__ a_src, const float* __restrict__ a_dst,
    float* __restrict__ g) {
  const int b = blockIdx.x >> 3;
  const int hh = blockIdx.x & 7;
  __shared__ float sd[512], ss[512], e1[512], e2[512], f1[512], f2[512], rZ[512];
  const int t = threadIdx.x;
  for (int i = t; i < 512; i += 256) {
    float dv = a_dst[((size_t)b * 512 + i) * 8 + hh];
    float sv = a_src[((size_t)b * 512 + i) * 8 + hh];
    sd[i] = dv; ss[i] = sv;
    e1[i] = __expf(dv); e2[i] = __expf(0.2f * dv);
    f1[i] = __expf(sv); f2[i] = __expf(0.2f * sv);
  }
  __syncthreads();
  for (int i = t; i < 512; i += 256) {
    float dv = sd[i];
    float z1 = 0.f, z2 = 0.f;
    for (int j = 0; j < 512; ++j) {
      bool p = (dv + ss[j]) >= 0.f;
      z1 += p ? f1[j] : 0.f;
      z2 += p ? 0.f : f2[j];
    }
    rZ[i] = 1.0f / (e1[i] * z1 + e2[i] * z2);
  }
  __syncthreads();
  for (int i = t; i < 512; i += 256) {
    float r = rZ[i];
    e1[i] *= r; e2[i] *= r;
  }
  __syncthreads();
  const float scale = 1.0f / (512.0f * 8.0f);  // fold the (1/(N*H)) mean here
  for (int j = t; j < 512; j += 256) {
    float sv = ss[j];
    float g1 = 0.f, g2 = 0.f;
    for (int i = 0; i < 512; ++i) {
      bool p = (sd[i] + sv) >= 0.f;
      g1 += p ? e1[i] : 0.f;
      g2 += p ? 0.f : e2[i];
    }
    g[((size_t)b * 8 + hh) * 512 + j] = (f1[j] * g1 + f2[j] * g2) * scale;
  }
}

// ---------------------------------------------------------------------------
// partial[b,chunk,c] = sum_{j in chunk} sum_h g[b,h,j]*h[b,j,h,c]
// ---------------------------------------------------------------------------
__global__ __launch_bounds__(256) void weighted_agg(
    const bf16_t* __restrict__ h, const float* __restrict__ g,
    float* __restrict__ partial) {
  const int b = blockIdx.y;
  const int chunk = blockIdx.x;
  const int c = threadIdx.x;
  float acc = 0.f;
  for (int jj = 0; jj < 64; ++jj) {
    const int j = chunk * 64 + jj;
    const bf16_t* hp = h + ((size_t)(b * 512 + j)) * 2048 + c;
    const float* gp = g + (size_t)b * 8 * 512 + j;
#pragma unroll
    for (int hh = 0; hh < 8; ++hh)
      acc += gp[hh * 512] * (float)hp[hh * 256];
  }
  partial[((size_t)b * 8 + chunk) * 256 + c] = acc;
}

__global__ __launch_bounds__(256) void finalize(
    const float* __restrict__ partial, const float* __restrict__ bias_g,
    float* __restrict__ out) {
  const int b = blockIdx.x;
  const int c = threadIdx.x;
  float acc = bias_g[c];
#pragma unroll
  for (int k = 0; k < 8; ++k) acc += partial[((size_t)b * 8 + k) * 256 + c];
  out[b * 256 + c] = acc;
}

// ---------------------------------------------------------------------------
extern "C" void kernel_launch(void* const* d_in, const int* in_sizes, int n_in,
                              void* d_out, int out_size, void* d_ws, size_t ws_size,
                              hipStream_t stream) {
  (void)in_sizes; (void)n_in; (void)out_size; (void)ws_size;
  const float* x       = (const float*)d_in[0];  // [64,512,768]
  const float* W1      = (const float*)d_in[1];  // [768,256]
  const float* b1      = (const float*)d_in[2];  // [256]
  const float* Wg      = (const float*)d_in[3];  // [256,2048]
  const float* att_src = (const float*)d_in[4];  // [8,256]
  const float* att_dst = (const float*)d_in[5];  // [8,256]
  const float* bias_g  = (const float*)d_in[6];  // [256]
  float* out = (float*)d_out;                    // [64,256]

  char* ws = (char*)d_ws;
  size_t off = 0;
  auto alloc = [&](size_t bytes) -> void* {
    void* p = ws + off;
    off = (off + bytes + 255) & ~(size_t)255;
    return p;
  };
  bf16_t* xb      = (bf16_t*)alloc(32768ull * 768 * 2);   //  50 MB
  bf16_t* W1t     = (bf16_t*)alloc(256ull * 768 * 2);     // [256,768]
  bf16_t* Wgt     = (bf16_t*)alloc(2048ull * 256 * 2);    // [2048,256]
  bf16_t* x1      = (bf16_t*)alloc(32768ull * 256 * 2);   //  16 MB
  bf16_t* h       = (bf16_t*)alloc(32768ull * 2048 * 2);  // 134 MB
  float*  a_src   = (float*)alloc(32768ull * 8 * 4);
  float*  a_dst   = (float*)alloc(32768ull * 8 * 4);
  float*  g       = (float*)alloc(64ull * 8 * 512 * 4);
  float*  partial = (float*)alloc(64ull * 8 * 256 * 4);

  // pre-passes: bf16 activations, bf16 transposed weights
  convert_bf16<<<(32768ull * 768) / (256 * 8), 256, 0, stream>>>(x, xb, 32768ll * 768);
  transpose_convert<<<dim3(256 / 64, 768 / 64), 256, 0, stream>>>(W1, W1t, 768, 256);
  transpose_convert<<<dim3(2048 / 64, 256 / 64), 256, 0, stream>>>(Wg, Wgt, 256, 2048);

  // GEMM1: x1 = xb @ W1 + b1   (M=32768, K=768, N=256)
  wmma_gemm<true><<<dim3(256 / 64, 32768 / 128), 256, 0, stream>>>(
      xb, W1t, b1, x1, 256, 768);
  // GEMM2: h = x1 @ Wg         (M=32768, K=256, N=2048)
  wmma_gemm<false><<<dim3(2048 / 64, 32768 / 128), 256, 0, stream>>>(
      x1, Wgt, nullptr, h, 2048, 256);

  attn_logits<<<32768, 256, 0, stream>>>(h, att_src, att_dst, a_src, a_dst);
  softmax_colsum<<<512, 256, 0, stream>>>(a_src, a_dst, g);
  weighted_agg<<<dim3(8, 64), 256, 0, stream>>>(h, g, partial);
  finalize<<<64, 256, 0, stream>>>(partial, bias_g, out);
}
